// RNNLayer_60790967108259
// MI455X (gfx1250) — compile-verified, matrix operand order checked by reading, not run on hardware
//
#include <hip/hip_runtime.h>
#include <hip/hip_bf16.h>

// ---------------------------------------------------------------------------
// Model constants (Griffin/Hawk block)
// ---------------------------------------------------------------------------
#define Hd 2048   // hidden
#define Rd 2048   // recurrent
#define Id 8192   // intermediate
#define NBd 8     // num blocks
#define BWd 256   // block width
#define CWd 4     // conv width
#define Bd 4
#define Td 2048
#define Md (Bd * Td)   // 8192 rows
#define CH (Bd * Rd)   // 8192 scan channels
#define SC 32          // scan chunks over T
#define LC (Td / SC)   // 64 steps per chunk

typedef __attribute__((ext_vector_type(16))) __bf16 v16bf;
typedef __attribute__((ext_vector_type(8)))  __bf16 v8bf;
typedef __attribute__((ext_vector_type(8)))  float  v8f;

// ---------------------------------------------------------------------------
// fp32 -> bf16 weight conversion
// ---------------------------------------------------------------------------
__global__ __launch_bounds__(256) void cvt_f32_bf16(const float* __restrict__ src,
                                                    __bf16* __restrict__ dst,
                                                    long n) {
  long i = (long)blockIdx.x * 256 + threadIdx.x;
  if (i < n) dst[i] = (__bf16)src[i];
}

// ---------------------------------------------------------------------------
// RMSNorm (C = 2048) -> bf16 output   (one block per row, 8 elems/thread)
// ---------------------------------------------------------------------------
__global__ __launch_bounds__(256) void rmsnorm_bf16(const float* __restrict__ x,
                                                    const float* __restrict__ w,
                                                    __bf16* __restrict__ out) {
  const long row = blockIdx.x;
  const float* xr = x + row * Hd;
  float vloc[8];
  float s = 0.f;
#pragma unroll
  for (int i = 0; i < 8; ++i) {
    int c = threadIdx.x + i * 256;
    float v = xr[c];
    vloc[i] = v;
    s += v * v;
  }
#pragma unroll
  for (int o = 16; o > 0; o >>= 1) s += __shfl_down(s, o, 32);
  __shared__ float red[8];
  if ((threadIdx.x & 31) == 0) red[threadIdx.x >> 5] = s;
  __syncthreads();
  __shared__ float rinv;
  if (threadIdx.x == 0) {
    float tot = 0.f;
#pragma unroll
    for (int i = 0; i < 8; ++i) tot += red[i];
    rinv = rsqrtf(tot / (float)Hd + 1e-6f);
  }
  __syncthreads();
  float ri = rinv;
#pragma unroll
  for (int i = 0; i < 8; ++i) {
    int c = threadIdx.x + i * 256;
    out[row * Hd + c] = (__bf16)(vloc[i] * ri * w[c]);
  }
}

// residual add + RMSNorm: sum_out(fp32) and norm_out(bf16)
__global__ __launch_bounds__(256) void add_rmsnorm_bf16(const float* __restrict__ resid,
                                                        const float* __restrict__ delta,
                                                        const float* __restrict__ w,
                                                        float* __restrict__ sum_out,
                                                        __bf16* __restrict__ norm_out) {
  const long row = blockIdx.x;
  float vloc[8];
  float s = 0.f;
#pragma unroll
  for (int i = 0; i < 8; ++i) {
    int c = threadIdx.x + i * 256;
    float v = resid[row * Hd + c] + delta[row * Hd + c];
    vloc[i] = v;
    sum_out[row * Hd + c] = v;
    s += v * v;
  }
#pragma unroll
  for (int o = 16; o > 0; o >>= 1) s += __shfl_down(s, o, 32);
  __shared__ float red[8];
  if ((threadIdx.x & 31) == 0) red[threadIdx.x >> 5] = s;
  __syncthreads();
  __shared__ float rinv;
  if (threadIdx.x == 0) {
    float tot = 0.f;
#pragma unroll
    for (int i = 0; i < 8; ++i) tot += red[i];
    rinv = rsqrtf(tot / (float)Hd + 1e-6f);
  }
  __syncthreads();
  float ri = rinv;
#pragma unroll
  for (int i = 0; i < 8; ++i) {
    int c = threadIdx.x + i * 256;
    norm_out[row * Hd + c] = (__bf16)(vloc[i] * ri * w[c]);
  }
}

// ---------------------------------------------------------------------------
// Tiled bf16 WMMA GEMM:  C[M,N] = A[M,K] * W[N,K]^T (+ bias[n])
// 128x128x32 tile, 256 threads = 8 waves, each wave owns 32(M) x 64(N).
// A-fragment layout (ISA 7.12.2, 16-bit A 16x32): lane L -> row L&15,
//   K in [kb,kb+8) and [16+kb,16+kb+8), kb=(L>>4)*8.  B loaded symmetrically
//   from row-major W[N,K].  C/D f32: reg v -> M = v + 8*(L>=16), N = L&15.
// ---------------------------------------------------------------------------
#define LROW 40  // 32 bf16 + 8 pad (80B rows, 16B aligned sub-chunks)

template <bool OUT_BF16>
__global__ __launch_bounds__(256) void gemm_wmma_bf16(
    const __bf16* __restrict__ A, int lda, long strideA,
    const __bf16* __restrict__ W, int ldw, long strideW,
    void* __restrict__ Cout, int ldc, long strideC,
    const float* __restrict__ bias, long strideBias,
    int M, int N, int K) {
  const int bz = blockIdx.z;
  A += (long)bz * strideA;
  W += (long)bz * strideW;
  const float* biasp = bias ? bias + (long)bz * strideBias : nullptr;
  char* Cb = (char*)Cout;
  Cb += (long)bz * strideC * (OUT_BF16 ? 2 : 4);

  __shared__ __bf16 Alds[2][128 * LROW];
  __shared__ __bf16 Blds[2][128 * LROW];

  const int t = threadIdx.x;
  const int lane = t & 31;
  const int wave = t >> 5;
  const int wm = wave & 3;   // 0..3  -> M offset wm*32
  const int wn = wave >> 2;  // 0..1  -> N offset wn*64
  const int r = lane & 15;
  const int kb = (lane >> 4) * 8;

  const int m0 = blockIdx.y * 128;
  const int n0 = blockIdx.x * 128;

  auto load_tiles = [&](int buf, int k0) {
#pragma unroll
    for (int it = 0; it < 2; ++it) {
      int e = it * 256 + t;       // 0..511
      int row = e >> 2;           // 0..127
      int col = (e & 3) << 3;     // 0,8,16,24
      *(v8bf*)&Alds[buf][row * LROW + col] =
          *(const v8bf*)(A + (long)(m0 + row) * lda + k0 + col);
      *(v8bf*)&Blds[buf][row * LROW + col] =
          *(const v8bf*)(W + (long)(n0 + row) * ldw + k0 + col);
    }
  };
  auto fragA = [&](int cur, int rrow) {
    union { v16bf v; v8bf h[2]; } u;
    const __bf16* p = &Alds[cur][rrow * LROW];
    u.h[0] = *(const v8bf*)(p + kb);
    u.h[1] = *(const v8bf*)(p + 16 + kb);
    return u.v;
  };
  auto fragB = [&](int cur, int rrow) {
    union { v16bf v; v8bf h[2]; } u;
    const __bf16* p = &Blds[cur][rrow * LROW];
    u.h[0] = *(const v8bf*)(p + kb);
    u.h[1] = *(const v8bf*)(p + 16 + kb);
    return u.v;
  };

  v8f acc[2][4];
#pragma unroll
  for (int mi = 0; mi < 2; ++mi)
#pragma unroll
    for (int nj = 0; nj < 4; ++nj) acc[mi][nj] = (v8f)0.f;

  const int nk = K >> 5;
  load_tiles(0, 0);
  __syncthreads();

  for (int kt = 0; kt < nk; ++kt) {
    const int cur = kt & 1;
    if (kt + 1 < nk) {
      load_tiles(cur ^ 1, (kt + 1) << 5);
      if (kt + 2 < nk) {  // L2-side prefetch of the tile after next
        __builtin_prefetch(A + (long)(m0 + (t >> 1)) * lda + ((kt + 2) << 5), 0, 1);
        __builtin_prefetch(W + (long)(n0 + (t >> 1)) * ldw + ((kt + 2) << 5), 0, 1);
      }
    }
    v16bf af[2], bfr[4];
#pragma unroll
    for (int mi = 0; mi < 2; ++mi) af[mi] = fragA(cur, wm * 32 + mi * 16 + r);
#pragma unroll
    for (int nj = 0; nj < 4; ++nj) bfr[nj] = fragB(cur, wn * 64 + nj * 16 + r);
#pragma unroll
    for (int mi = 0; mi < 2; ++mi)
#pragma unroll
      for (int nj = 0; nj < 4; ++nj)
        acc[mi][nj] = __builtin_amdgcn_wmma_f32_16x16x32_bf16(
            false, af[mi], false, bfr[nj], (short)0, acc[mi][nj], false, false);
    __syncthreads();
  }

  // epilogue
  const int radd = (lane >> 4) * 8;
  const int ccol0 = n0 + wn * 64 + (lane & 15);
#pragma unroll
  for (int mi = 0; mi < 2; ++mi) {
#pragma unroll
    for (int nj = 0; nj < 4; ++nj) {
      int col = ccol0 + nj * 16;
      float bv = biasp ? biasp[col] : 0.f;
#pragma unroll
      for (int v = 0; v < 8; ++v) {
        int row = m0 + wm * 32 + mi * 16 + radd + v;
        float val = acc[mi][nj][v] + bv;
        if (OUT_BF16)
          ((__bf16*)Cb)[(long)row * ldc + col] = (__bf16)val;
        else
          ((float*)Cb)[(long)row * ldc + col] = val;
      }
    }
  }
}

// ---------------------------------------------------------------------------
// Depthwise causal conv over T (width 4); xy is [M, 2R], uses first R cols.
// Writes fp32 xc and bf16 copy for the block-diagonal WMMA GEMMs.
// ---------------------------------------------------------------------------
__global__ __launch_bounds__(256) void conv_kernel(const float* __restrict__ xy,
                                                   const float* __restrict__ cw,
                                                   const float* __restrict__ cb,
                                                   float* __restrict__ xc,
                                                   __bf16* __restrict__ xcbf) {
  long i = (long)blockIdx.x * 256 + threadIdx.x;  // over B*T*R
  int r = (int)(i & (Rd - 1));
  long bt = i >> 11;  // /Rd
  int tt = (int)(bt & (Td - 1));
  int b = (int)(bt >> 11);
  float acc = cb[r];
#pragma unroll
  for (int k = 0; k < CWd; ++k) {
    int ts = tt + k - (CWd - 1);
    if (ts >= 0)
      acc += cw[k * Rd + r] * xy[((long)(b * Td + ts)) * (2 * Rd) + r];
  }
  xc[i] = acc;
  xcbf[i] = (__bf16)acc;
}

// ---------------------------------------------------------------------------
// RG-LRU chunked scan (3 passes; 32 chunks of 64 steps over T).
// Pass 1: compute per-step (a, b), overwrite x_rg/a_rg buffers with them,
//         emit per-chunk aggregates (prod(a), h_local).
// ---------------------------------------------------------------------------
__global__ __launch_bounds__(256) void scan_pass1(const float* __restrict__ xc,
                                                  float* __restrict__ xrg,  // in: x_rg pre, out: a
                                                  float* __restrict__ arg_, // in: a_rg pre, out: b
                                                  const float* __restrict__ ap,
                                                  float* __restrict__ chunk_a,
                                                  float* __restrict__ chunk_h) {
  int idx = blockIdx.x * 256 + threadIdx.x;  // chunk*CH + ch
  int ch = idx & (CH - 1);
  int chunk = idx >> 13;                     // / CH
  int r = ch & (Rd - 1);
  int b = ch >> 11;
  float sp = log1pf(expf(ap[r]));  // softplus(a_param)
  float h = 0.f, aprod = 1.f;
  long rowbase = (long)b * Td + chunk * LC;
  for (int t = 0; t < LC; ++t) {
    long i = (rowbase + t) * Rd + r;
    float gx = 1.f / (1.f + expf(-xrg[i]));
    float ga = 1.f / (1.f + expf(-arg_[i]));
    float la = -8.f * ga * sp;
    float a = expf(la);
    float mult = sqrtf(fmaxf(0.f, 1.f - expf(2.f * la)));
    float bv = xc[i] * gx * mult;
    xrg[i] = a;
    arg_[i] = bv;
    h = a * h + bv;
    aprod *= a;
  }
  chunk_a[idx] = aprod;
  chunk_h[idx] = h;
}

// Pass 2: per-channel sequential scan over 32 chunk aggregates; chunk_h is
// overwritten with the *incoming* prefix state for each chunk.
__global__ __launch_bounds__(256) void scan_pass2(const float* __restrict__ chunk_a,
                                                  float* __restrict__ chunk_h) {
  int ch = blockIdx.x * 256 + threadIdx.x;  // 0..CH-1
  float h = 0.f;
  for (int c = 0; c < SC; ++c) {
    int i = c * CH + ch;
    float a = chunk_a[i];
    float he = chunk_h[i];
    chunk_h[i] = h;  // incoming prefix for chunk c
    h = a * h + he;
  }
}

// Pass 3: replay h = a*h + b from the chunk prefix; fused gelu(y)*h -> bf16.
__global__ __launch_bounds__(256) void scan_pass3(const float* __restrict__ abuf,
                                                  const float* __restrict__ bbuf,
                                                  const float* __restrict__ chunk_h,
                                                  const float* __restrict__ xy,
                                                  __bf16* __restrict__ gy) {
  int idx = blockIdx.x * 256 + threadIdx.x;  // chunk*CH + ch
  int ch = idx & (CH - 1);
  int chunk = idx >> 13;
  int r = ch & (Rd - 1);
  int b = ch >> 11;
  float h = chunk_h[idx];
  long rowbase = (long)b * Td + chunk * LC;
  for (int t = 0; t < LC; ++t) {
    long i = (rowbase + t) * Rd + r;
    h = abuf[i] * h + bbuf[i];
    float yv = xy[(rowbase + t) * (2 * Rd) + Rd + r];
    float ge = 0.5f * yv * (1.f + erff(yv * 0.70710678118654752f));
    gy[i] = (__bf16)(ge * h);
  }
}

// ---------------------------------------------------------------------------
// MLP activation: act = gelu(gate) * up   (gu is [M, 2I] bf16)
// ---------------------------------------------------------------------------
__global__ __launch_bounds__(256) void gelu_mul_kernel(const __bf16* __restrict__ gu,
                                                       __bf16* __restrict__ act) {
  long i = (long)blockIdx.x * 256 + threadIdx.x;  // over M*I
  long row = i >> 13;                             // /Id
  int n = (int)(i & (Id - 1));
  float g = (float)gu[row * (2 * Id) + n];
  float u = (float)gu[row * (2 * Id) + Id + n];
  float ge = 0.5f * g * (1.f + erff(g * 0.70710678118654752f));
  act[i] = (__bf16)(ge * u);
}

__global__ __launch_bounds__(256) void add_kernel(const float* __restrict__ a,
                                                  const float* __restrict__ b,
                                                  float* __restrict__ o, long n) {
  long i = (long)blockIdx.x * 256 + threadIdx.x;
  if (i < n) o[i] = a[i] + b[i];
}

// ---------------------------------------------------------------------------
// Host launcher
// ---------------------------------------------------------------------------
extern "C" void kernel_launch(void* const* d_in, const int* in_sizes, int n_in,
                              void* d_out, int out_size, void* d_ws, size_t ws_size,
                              hipStream_t stream) {
  const float* hidden  = (const float*)d_in[0];
  const float* w_xy    = (const float*)d_in[1];
  const float* conv_w  = (const float*)d_in[2];
  const float* conv_b  = (const float*)d_in[3];
  const float* bd_in_w = (const float*)d_in[4];
  const float* bd_in_b = (const float*)d_in[5];
  const float* bd_a_w  = (const float*)d_in[6];
  const float* bd_a_b  = (const float*)d_in[7];
  const float* a_param = (const float*)d_in[8];
  const float* w_rp    = (const float*)d_in[9];
  const float* w_gu    = (const float*)d_in[10];
  const float* w_mr    = (const float*)d_in[11];
  const float* norm1   = (const float*)d_in[12];
  const float* norm2   = (const float*)d_in[13];
  float* out = (float*)d_out;

  // ---- workspace carve-out ----
  char* p = (char*)d_ws;
  auto alloc = [&](size_t bytes) -> void* {
    void* q = (void*)p;
    p += (bytes + 255) & ~(size_t)255;
    return q;
  };
  __bf16* wxy_bf = (__bf16*)alloc((size_t)2 * Rd * Hd * 2);       // [4096,2048]
  __bf16* win_bf = (__bf16*)alloc((size_t)NBd * BWd * BWd * 2);
  __bf16* wa_bf  = (__bf16*)alloc((size_t)NBd * BWd * BWd * 2);
  __bf16* wrp_bf = (__bf16*)alloc((size_t)Hd * Rd * 2);
  __bf16* wgu_bf = (__bf16*)alloc((size_t)2 * Id * Hd * 2);
  __bf16* wmr_bf = (__bf16*)alloc((size_t)Hd * Id * 2);
  __bf16* hsn_bf = (__bf16*)alloc((size_t)Md * Hd * 2);
  float*  xy_f   = (float*) alloc((size_t)Md * 2 * Rd * 4);
  float*  xc_f   = (float*) alloc((size_t)Md * Rd * 4);
  __bf16* xc_bf  = (__bf16*)alloc((size_t)Md * Rd * 2);
  float*  xrg_f  = (float*) alloc((size_t)Md * Rd * 4);   // reused as a-buffer
  float*  arg_f  = (float*) alloc((size_t)Md * Rd * 4);   // reused as b-buffer
  __bf16* gy_bf  = (__bf16*)alloc((size_t)Md * Rd * 2);
  float*  recc_f = (float*) alloc((size_t)Md * Hd * 4);
  float*  hs2_f  = (float*) alloc((size_t)Md * Hd * 4);
  __bf16* hs2n_bf= (__bf16*)alloc((size_t)Md * Hd * 2);
  __bf16* gu_bf  = (__bf16*)alloc((size_t)Md * 2 * Id * 2);
  __bf16* act_bf = (__bf16*)alloc((size_t)Md * Id * 2);
  float*  mlp_f  = (float*) alloc((size_t)Md * Hd * 4);
  float*  cka_f  = (float*) alloc((size_t)SC * CH * 4);   // chunk prod(a)
  float*  ckh_f  = (float*) alloc((size_t)SC * CH * 4);   // chunk h / prefix

  auto cvt = [&](const float* s, __bf16* d, long n) {
    cvt_f32_bf16<<<(unsigned)((n + 255) / 256), 256, 0, stream>>>(s, d, n);
  };
  // ---- weight conversion (weights then live in L2: 192MB >> 128MB of bf16) ----
  cvt(w_xy,    wxy_bf, (long)2 * Rd * Hd);
  cvt(bd_in_w, win_bf, (long)NBd * BWd * BWd);
  cvt(bd_a_w,  wa_bf,  (long)NBd * BWd * BWd);
  cvt(w_rp,    wrp_bf, (long)Hd * Rd);
  cvt(w_gu,    wgu_bf, (long)2 * Id * Hd);
  cvt(w_mr,    wmr_bf, (long)Hd * Id);

  // ---- norm1 ----
  rmsnorm_bf16<<<Md, 256, 0, stream>>>(hidden, norm1, hsn_bf);

  // ---- xy = hs_n @ W_xy^T : [8192,4096] ----
  gemm_wmma_bf16<false><<<dim3(2 * Rd / 128, Md / 128, 1), 256, 0, stream>>>(
      hsn_bf, Hd, 0, wxy_bf, Hd, 0, xy_f, 2 * Rd, 0, nullptr, 0, Md, 2 * Rd, Hd);

  // ---- causal conv ----
  conv_kernel<<<(unsigned)(((long)Md * Rd) / 256), 256, 0, stream>>>(
      xy_f, conv_w, conv_b, xc_f, xc_bf);

  // ---- block-diagonal gates (batched over 8 blocks) ----
  gemm_wmma_bf16<false><<<dim3(BWd / 128, Md / 128, NBd), 256, 0, stream>>>(
      xc_bf, Rd, BWd, win_bf, BWd, (long)BWd * BWd,
      xrg_f, Rd, BWd, bd_in_b, BWd, Md, BWd, BWd);
  gemm_wmma_bf16<false><<<dim3(BWd / 128, Md / 128, NBd), 256, 0, stream>>>(
      xc_bf, Rd, BWd, wa_bf, BWd, (long)BWd * BWd,
      arg_f, Rd, BWd, bd_a_b, BWd, Md, BWd, BWd);

  // ---- RG-LRU chunked scan (3 passes), fused gelu(y)*h in pass 3 ----
  scan_pass1<<<(SC * CH) / 256, 256, 0, stream>>>(xc_f, xrg_f, arg_f, a_param,
                                                  cka_f, ckh_f);
  scan_pass2<<<CH / 256, 256, 0, stream>>>(cka_f, ckh_f);
  scan_pass3<<<(SC * CH) / 256, 256, 0, stream>>>(xrg_f, arg_f, ckh_f, xy_f,
                                                  gy_bf);

  // ---- recurrent down-projection ----
  gemm_wmma_bf16<false><<<dim3(Hd / 128, Md / 128, 1), 256, 0, stream>>>(
      gy_bf, Rd, 0, wrp_bf, Rd, 0, recc_f, Hd, 0, nullptr, 0, Md, Hd, Rd);

  // ---- residual add + norm2 ----
  add_rmsnorm_bf16<<<Md, 256, 0, stream>>>(hidden, recc_f, norm2, hs2_f, hs2n_bf);

  // ---- gate-up projection (bf16 output to halve traffic) ----
  gemm_wmma_bf16<true><<<dim3(2 * Id / 128, Md / 128, 1), 256, 0, stream>>>(
      hs2n_bf, Hd, 0, wgu_bf, Hd, 0, gu_bf, 2 * Id, 0, nullptr, 0, Md, 2 * Id, Hd);

  // ---- gelu(gate)*up ----
  gelu_mul_kernel<<<(unsigned)(((long)Md * Id) / 256), 256, 0, stream>>>(gu_bf,
                                                                         act_bf);

  // ---- MLP down-projection ----
  gemm_wmma_bf16<false><<<dim3(Hd / 128, Md / 128, 1), 256, 0, stream>>>(
      act_bf, Id, 0, wmr_bf, Id, 0, mlp_f, Hd, 0, nullptr, 0, Md, Hd, Id);

  // ---- final residual ----
  add_kernel<<<(unsigned)(((long)Md * Hd) / 256), 256, 0, stream>>>(
      hs2_f, mlp_f, out, (long)Md * Hd);
}
